// Rotatory_Attention_Unet_49400713838929
// MI455X (gfx1250) — matrix operand order repaired and checked by reading, not compiled, not run on hardware
//
#include <hip/hip_runtime.h>
#include <hip/hip_bf16.h>

#define DEV __device__ __forceinline__

typedef __attribute__((ext_vector_type(16))) __bf16 v16bf;
typedef __attribute__((ext_vector_type(8)))  __bf16 v8bf;
typedef __attribute__((ext_vector_type(4)))  __bf16 v4bf;
typedef __attribute__((ext_vector_type(8)))  float  v8f;
typedef __attribute__((ext_vector_type(4)))  float  v4f;
typedef __attribute__((ext_vector_type(4)))  int    v4i;

// ---------- async global->LDS path (guarded; falls back to plain copy) ------
#if defined(__has_builtin)
#if __has_builtin(__builtin_amdgcn_global_load_async_to_lds_b128) && \
    __has_builtin(__builtin_amdgcn_s_wait_asynccnt)
#define RAU_HAS_ASYNC 1
#endif
#endif
#ifndef RAU_HAS_ASYNC
#define RAU_HAS_ASYNC 0
#endif
typedef __attribute__((address_space(1))) v4i as1_v4i;  // global int4
typedef __attribute__((address_space(3))) v4i as3_v4i;  // LDS int4

// ---------- bf16 helpers (native convert, RNE) ----------
DEV __bf16 f2bf(float f) { return (__bf16)f; }
DEV unsigned short f2bfbits(float f) {
  return __builtin_bit_cast(unsigned short, (__bf16)f);
}

// ---------- WMMA core (CDNA5 wave32, 16x16x32 bf16 -> f32) ----------
DEV v8f wmma_bf16(v16bf a, v16bf b, v8f c) {
  return __builtin_amdgcn_wmma_f32_16x16x32_bf16(false, a, false, b, (short)0, c,
                                                 false, false);
}

DEV v16bf cat16(v8bf lo, v8bf hi) {
  v16bf r;
#pragma unroll
  for (int i = 0; i < 8; ++i) {
    r[i] = lo[i];
    r[8 + i] = hi[i];
  }
  return r;
}

// A fragment 16x32 from row-major [m][k] (row stride rs halfwords), tile at k0.
// Per-lane K runs are contiguous: {hi*8..hi*8+7} and {16+hi*8..16+hi*8+7}.
DEV v16bf frag_a_rs(const __bf16* A, int rs, int lane, int k0) {
  const int m = lane & 15, hi = (lane >> 4) & 1;
  const __bf16* p = A + (long)m * rs + k0;
  return cat16(*(const v8bf*)(p + hi * 8), *(const v8bf*)(p + 16 + hi * 8));
}
// B fragment 32x16 where element (k,n) = B[n*rs + k] (k contiguous per column).
// Per-lane K runs are contiguous: {hi*8..hi*8+7} and {16+hi*8..16+hi*8+7}.
DEV v16bf frag_b_rs(const __bf16* B, long rs, int lane, long k0) {
  const int n = lane & 15, hi = (lane >> 4) & 1;
  const __bf16* p = B + (long)n * rs + k0 + hi * 8;
  return cat16(*(const v8bf*)p, *(const v8bf*)(p + 16));
}

// =====================================================================
// Conv 3x3 pad 1 (+bias) as implicit GEMM, LDS-staged tiles.
// block 256 = 8 waves; macro tile 32(oc) x 64(pixels); wave -> 16x16 tile.
// =====================================================================
__global__ __launch_bounds__(256) void rau_conv3x3_kernel(
    const float* __restrict__ in, const float* __restrict__ wgt,
    const float* __restrict__ bias, float* __restrict__ out, int NB, int IC,
    int OC, int H, int W) {
  __shared__ alignas(16) __bf16 As[32 * 32];
  __shared__ alignas(16) __bf16 Bs[64 * 32];
  const int HW = H * W;
  const int K = IC * 9;
  const int tid = threadIdx.x, lane = tid & 31, wave = tid >> 5;
  const int ocBase = blockIdx.y * 32, colBase = blockIdx.x * 64;
  const int wm = (wave >> 2) * 16, wn = (wave & 3) * 16;

  // A staging: thread -> (row am, 4 consecutive k)
  const int am = tid >> 3, ak = (tid & 7) * 4;
  const float* arow = wgt + (long)(ocBase + am) * K;
  const bool k4 = (K & 3) == 0;
  // B staging: thread -> (col bn, 8 consecutive k), branchless im2col
  const int bn = tid >> 2, bk = (tid & 3) * 8;
  const int j = colBase + bn;
  const int nb = j / HW, p = j - nb * HW;
  const int py = p / W, px = p - py * W;
  const float* inb = in + (long)nb * IC * HW;

  v8f acc = {};
  for (int k0 = 0; k0 < K; k0 += 32) {
    if (k4 && (k0 + 32) <= K) {  // fast path: aligned b128 weight load
      v4f wv = *(const v4f*)(arow + k0 + ak);
      v4bf t;
#pragma unroll
      for (int i = 0; i < 4; ++i) t[i] = f2bf(wv[i]);
      *(v4bf*)&As[am * 32 + ak] = t;
    } else {
      v4bf t;
#pragma unroll
      for (int i = 0; i < 4; ++i) {
        int k = k0 + ak + i;
        int ks = k < K ? k : 0;
        float v = arow[ks];
        t[i] = f2bf(k < K ? v : 0.f);
      }
      *(v4bf*)&As[am * 32 + ak] = t;
    }
    {
      v8bf t;
#pragma unroll
      for (int i = 0; i < 8; ++i) {
        int k = k0 + bk + i;
        int ks = k < K ? k : 0;
        int ic = ks / 9, r = ks - ic * 9;
        int ky = r / 3, kx = r - ky * 3;
        int y = py + ky - 1, x = px + kx - 1;
        bool ok = (k < K) && ((unsigned)y < (unsigned)H) &&
                  ((unsigned)x < (unsigned)W);
        long idx = (long)ic * HW + (long)(ok ? y : 0) * W + (ok ? x : 0);
        float v = inb[idx];
        t[i] = f2bf(ok ? v : 0.f);
      }
      *(v8bf*)&Bs[bn * 32 + bk] = t;
    }
    __syncthreads();
    acc = wmma_bf16(frag_a_rs(As + wm * 32, 32, lane, 0),
                    frag_b_rs(Bs + wn * 32, 32, lane, 0), acc);
    __syncthreads();
  }
  const int hi = (lane >> 4) & 1;
  const int jj = colBase + wn + (lane & 15);
  const int onb = jj / HW, op = jj - onb * HW;
#pragma unroll
  for (int r = 0; r < 8; ++r) {
    int oc = ocBase + wm + r + hi * 8;
    out[((long)onb * OC + oc) * HW + op] = acc[r] + bias[oc];
  }
}

// =====================================================================
// BatchNorm training-mode stats + apply
// =====================================================================
__global__ __launch_bounds__(256) void rau_bn_stats_kernel(
    const float* __restrict__ x, float* __restrict__ mean,
    float* __restrict__ rstd, int NB, int C, int HW) {
  const int c = blockIdx.x;
  const long cnt = (long)NB * HW;
  float s = 0.f, s2 = 0.f;
  for (long t = threadIdx.x; t < cnt; t += 256) {
    long n = t / HW, p = t - n * HW;
    float v = x[(n * C + c) * HW + p];
    s += v;
    s2 += v * v;
  }
  __shared__ float sh[256], sh2[256];
  sh[threadIdx.x] = s;
  sh2[threadIdx.x] = s2;
  __syncthreads();
  for (int st = 128; st > 0; st >>= 1) {
    if ((int)threadIdx.x < st) {
      sh[threadIdx.x] += sh[threadIdx.x + st];
      sh2[threadIdx.x] += sh2[threadIdx.x + st];
    }
    __syncthreads();
  }
  if (threadIdx.x == 0) {
    float m = sh[0] / (float)cnt;
    float var = sh2[0] / (float)cnt - m * m;
    mean[c] = m;
    rstd[c] = rsqrtf(var + 1e-5f);
  }
}

__global__ void rau_bn_relu_kernel(const float* __restrict__ x,
                                   const float* __restrict__ mean,
                                   const float* __restrict__ rstd,
                                   const float* __restrict__ g,
                                   const float* __restrict__ beta,
                                   float* __restrict__ y, int C, int HW,
                                   long total) {
  for (long i = (long)blockIdx.x * blockDim.x + threadIdx.x; i < total;
       i += (long)gridDim.x * blockDim.x) {
    long c = (i / HW) % C;
    float v = (x[i] - mean[c]) * rstd[c] * g[c] + beta[c];
    y[i] = v > 0.f ? v : 0.f;
  }
}

__global__ void rau_maxpool_kernel(const float* __restrict__ x,
                                   float* __restrict__ y, long NC, int H,
                                   int W) {
  const int Ho = H >> 1, Wo = W >> 1;
  const long tot = NC * Ho * Wo;
  for (long i = (long)blockIdx.x * blockDim.x + threadIdx.x; i < tot;
       i += (long)gridDim.x * blockDim.x) {
    long p = i % (Ho * Wo);
    long nc = i / (Ho * Wo);
    int yo = (int)(p / Wo), xo = (int)(p - (p / Wo) * Wo);
    const float* b = x + nc * H * W + (2 * yo) * W + 2 * xo;
    float m0 = b[0] > b[1] ? b[0] : b[1];
    float m1 = b[W] > b[W + 1] ? b[W] : b[W + 1];
    y[i] = m0 > m1 ? m0 : m1;
  }
}

__global__ void rau_copy_kernel(const float* __restrict__ src,
                                float* __restrict__ dst, long n) {
  for (long i = (long)blockIdx.x * blockDim.x + threadIdx.x; i < n;
       i += (long)gridDim.x * blockDim.x)
    dst[i] = src[i];
}

// =====================================================================
// Attention projections (LDS-staged GEMM): out = rows(x) @ Wp  (bf16 out)
//  nbrMode=0: rows of slice v+reach (Q).  nbrMode=1: stacked neighbors (K/V).
//  transOut=1 writes out[v][d][row] (transposed, used for V).
// =====================================================================
__global__ __launch_bounds__(256) void rau_attn_proj_kernel(
    const float* __restrict__ x, const float* __restrict__ wp,
    unsigned short* __restrict__ outbf, int D, int L, int M, int reach,
    int nbrMode, int transOut) {
  __shared__ alignas(16) __bf16 As[32 * 32];
  __shared__ alignas(16) __bf16 Bs[64 * 32];
  const int v = blockIdx.z;
  const int tid = threadIdx.x, lane = tid & 31, wave = tid >> 5;
  const int rowBase = blockIdx.y * 32, colBase = blockIdx.x * 64;
  const int wm = (wave >> 2) * 16, wn = (wave & 3) * 16;
  const long DL = (long)D * L;
  const int kk = tid >> 3;       // 0..31 : k within tile
  const int m4 = (tid & 7) * 4;  // A rows handled by this thread
  const int n8 = (tid & 7) * 8;  // B cols handled by this thread

  v8f acc = {};
  for (int k0 = 0; k0 < D; k0 += 32) {
    {  // stage A: x rows (l contiguous in memory), branchless row guard
      const int k = k0 + kk;
#pragma unroll
      for (int i = 0; i < 4; ++i) {
        int row = rowBase + m4 + i;
        bool ok = row < M;
        int rr = ok ? row : 0;
        int part = nbrMode ? rr / L : 0;
        int l = rr - part * L;
        int offp = (part < reach) ? (part - reach) : (part - reach + 1);
        int slice = v + reach + (nbrMode ? offp : 0);
        float val = x[(long)slice * DL + (long)k * L + l];
        As[(m4 + i) * 32 + kk] = f2bf(ok ? val : 0.f);
      }
    }
    {  // stage B: Wp row k, 8 consecutive cols -> two b128 loads
      const float* wr = wp + (long)(k0 + kk) * D + colBase + n8;
      v4f a0 = *(const v4f*)wr;
      v4f a1 = *(const v4f*)(wr + 4);
#pragma unroll
      for (int i = 0; i < 4; ++i) {
        Bs[(n8 + i) * 32 + kk] = f2bf(a0[i]);
        Bs[(n8 + 4 + i) * 32 + kk] = f2bf(a1[i]);
      }
    }
    __syncthreads();
    acc = wmma_bf16(frag_a_rs(As + wm * 32, 32, lane, 0),
                    frag_b_rs(Bs + wn * 32, 32, lane, 0), acc);
    __syncthreads();
  }
  const int hi = (lane >> 4) & 1, n = lane & 15;
#pragma unroll
  for (int r = 0; r < 8; ++r) {
    int row = rowBase + wm + r + hi * 8;
    int col = colBase + wn + n;
    if (row < M) {
      unsigned short b = f2bfbits(acc[r]);
      if (transOut)
        outbf[((long)v * D + col) * M + row] = b;
      else
        outbf[((long)v * M + row) * D + col] = b;
    }
  }
}

// =====================================================================
// Flash attention: WG per (16-row l-tile, valid slice v). Online softmax over
// M=4L keys in 32-chunks. Q tile arrives via ASYNC global->LDS DMA (when the
// toolchain exposes it), Q/P fragments from LDS (b128), K/V fragments read
// directly from global as contiguous b128 runs (V is pre-transposed [v][d][m]).
// =====================================================================
__global__ __launch_bounds__(128) void rau_flash_attn_kernel(
    const unsigned short* __restrict__ Q, const unsigned short* __restrict__ Kb,
    const unsigned short* __restrict__ VbT, float* __restrict__ nx, int D,
    int L, int M, int reach) {
  extern __shared__ float smem[];
  float* O = smem;            // 16*D f32
  float* S = O + 16 * D;      // 16*32 f32
  float* rowm = S + 512;      // 16
  float* rowl = rowm + 16;    // 16
  float* ralpha = rowl + 16;  // 16
  __bf16* Qs = (__bf16*)(ralpha + 16);  // 16*D bf16 (row-major, k contiguous)
  __bf16* P = Qs + 16 * D;              // 16*32 bf16

  const int tid = threadIdx.x, lane = tid & 31, wave = tid >> 5;
  const int v = blockIdx.y, lt = blockIdx.x;
  const long qbase = ((long)v * L + (long)lt * 16) * D;

#if RAU_HAS_ASYNC
  {
    // ASYNCcnt-tracked DMA of the 16xD bf16 Q tile: 16B per lane per issue.
    const unsigned short* gq = Q + qbase;
    const int iters = D >> 6;  // (16*D halfwords) / (128 lanes * 8 halfwords)
    for (int it = 0; it < iters; ++it) {
      int o16 = (it * 128 + tid) * 8;  // halfword offset, 16B aligned
      __builtin_amdgcn_global_load_async_to_lds_b128(
          (as1_v4i*)(unsigned long)(gq + o16),
          (as3_v4i*)(unsigned)(unsigned long)(Qs + o16), 0, 0);
    }
    for (int i = tid; i < 16 * D; i += 128) O[i] = 0.f;
    __builtin_amdgcn_s_wait_asynccnt(0);
  }
#else
  for (int i = tid; i < 16 * D; i += 128) {
    ((unsigned short*)Qs)[i] = Q[qbase + i];
    O[i] = 0.f;
  }
#endif
  if (tid < 16) {
    rowm[tid] = -3.0e38f;
    rowl[tid] = 0.f;
  }
  __syncthreads();

  const float scale = 1.0f / sqrtf((float)D);
  const long kvslice = (long)v * M;
  const int hi = (lane >> 4) & 1, nn = lane & 15;

  for (int mb = 0; mb < M; mb += 32) {
    if (mb + 32 < M)  // speculative prefetch of next K chunk
      __builtin_prefetch(
          (const void*)(Kb + (kvslice + mb + 32) * D + (long)tid * (D >> 2)), 0,
          1);
    // ---- S = scale * Q @ K^T (waves 0,1 -> 16 key-cols each) ----
    if (wave < 2) {
      const __bf16* Kt = (const __bf16*)(Kb + (kvslice + mb + wave * 16) * D);
      v8f acc = {};
      for (int k0 = 0; k0 < D; k0 += 32)
        acc = wmma_bf16(frag_a_rs(Qs, D, lane, k0), frag_b_rs(Kt, D, lane, k0),
                        acc);
#pragma unroll
      for (int r = 0; r < 8; ++r)
        S[(r + hi * 8) * 32 + wave * 16 + nn] = acc[r] * scale;
    }
    __syncthreads();
    // ---- online softmax update (one thread per row) ----
    if (tid < 16) {
      float mOld = rowm[tid];
      float mNew = mOld;
#pragma unroll
      for (int jx = 0; jx < 32; ++jx) {
        float s = S[tid * 32 + jx];
        mNew = s > mNew ? s : mNew;
      }
      float a = __expf(mOld - mNew);
      float ssum = 0.f;
#pragma unroll
      for (int jx = 0; jx < 32; ++jx) {
        float pv = __expf(S[tid * 32 + jx] - mNew);
        P[tid * 32 + jx] = f2bf(pv);
        ssum += pv;
      }
      rowl[tid] = rowl[tid] * a + ssum;
      rowm[tid] = mNew;
      ralpha[tid] = a;
    }
    __syncthreads();
    // ---- O = O*alpha + P @ V_chunk, d-column tiles split over 4 waves ----
    for (int ct = wave; ct < (D >> 4); ct += 4) {
      const __bf16* Vt = (const __bf16*)(VbT + ((long)v * D + ct * 16) * M + mb);
      v8f c;
#pragma unroll
      for (int r = 0; r < 8; ++r)
        c[r] = O[(r + hi * 8) * D + ct * 16 + nn] * ralpha[r + hi * 8];
      c = wmma_bf16(frag_a_rs(P, 32, lane, 0), frag_b_rs(Vt, M, lane, 0), c);
#pragma unroll
      for (int r = 0; r < 8; ++r) O[(r + hi * 8) * D + ct * 16 + nn] = c[r];
    }
    __syncthreads();
  }

  // finalize + scrambled write (reference view(L,D) -> (D,H*W))
  const long sliceBase = (long)(v + reach) * D * L;
  for (int i = tid; i < 16 * D; i += 128) {
    int r = i / D, d = i - (i / D) * D;
    long f = (long)(lt * 16 + r) * D + d;
    long c = f / L, p = f - (f / L) * L;
    nx[sliceBase + c * L + p] = O[i] / rowl[r];
  }
}

// =====================================================================
// ConvTranspose2d k=2 s=2 as 4 parity GEMMs (LDS-staged), fused skip+relu+
// sigmoid; writes channels [0,OC) of the concat buffer (2*OC channels).
// =====================================================================
__global__ __launch_bounds__(256) void rau_deconv_fuse_kernel(
    const float* __restrict__ xin, const float* __restrict__ upw,
    const float* __restrict__ upb, const float* __restrict__ skip,
    float* __restrict__ cat, int NB, int C, int OC, int h, int w) {
  __shared__ alignas(16) __bf16 As[32 * 32];
  __shared__ alignas(16) __bf16 Bs[64 * 32];
  const int pp = blockIdx.z;
  const int ky = pp >> 1, kx = pp & 1;
  const int hw = h * w, W2 = 2 * w, HW2 = 4 * hw;
  const int tid = threadIdx.x, lane = tid & 31, wave = tid >> 5;
  const int ocBase = blockIdx.y * 32, colBase = blockIdx.x * 64;
  const int wm = (wave >> 2) * 16, wn = (wave & 3) * 16;

  const int am = tid >> 3, ak = (tid & 7) * 4;
  const int bn = tid >> 2, bk = (tid & 3) * 8;
  const int j = colBase + bn;
  const int nb = j / hw, p = j - nb * hw;
  const float* xb = xin + (long)nb * C * hw + p;

  v8f acc = {};
  for (int k0 = 0; k0 < C; k0 += 32) {
    {
      v4bf t;
#pragma unroll
      for (int i = 0; i < 4; ++i)
        t[i] = f2bf(upw[((long)(k0 + ak + i) * OC + ocBase + am) * 4 + pp]);
      *(v4bf*)&As[am * 32 + ak] = t;
    }
    {
      v8bf t;
#pragma unroll
      for (int i = 0; i < 8; ++i) t[i] = f2bf(xb[(long)(k0 + bk + i) * hw]);
      *(v8bf*)&Bs[bn * 32 + bk] = t;
    }
    __syncthreads();
    acc = wmma_bf16(frag_a_rs(As + wm * 32, 32, lane, 0),
                    frag_b_rs(Bs + wn * 32, 32, lane, 0), acc);
    __syncthreads();
  }
  const int hi = (lane >> 4) & 1;
  const int jj = colBase + wn + (lane & 15);
  const int nb2 = jj / hw, p2 = jj - nb2 * hw;
  const int yy = p2 / w, xx = p2 - yy * w;
  const int q2 = (2 * yy + ky) * W2 + 2 * xx + kx;
#pragma unroll
  for (int r = 0; r < 8; ++r) {
    int oc = ocBase + wm + r + hi * 8;
    float val = acc[r] + upb[oc] + skip[((long)nb2 * OC + oc) * HW2 + q2];
    val = val > 0.f ? val : 0.f;       // relu
    val = 1.f / (1.f + __expf(-val));  // sigmoid
    cat[((long)nb2 * (2 * OC) + oc) * HW2 + q2] = val;
  }
}

__global__ void rau_concat_skip_kernel(const float* __restrict__ skip,
                                       float* __restrict__ cat, long NB,
                                       long OC, long HW2) {
  const long tot = NB * OC * HW2;
  for (long i = (long)blockIdx.x * blockDim.x + threadIdx.x; i < tot;
       i += (long)gridDim.x * blockDim.x) {
    long p = i % HW2;
    long t = i / HW2;
    long c = t % OC;
    long nb = t / OC;
    cat[(nb * 2 * OC + OC + c) * HW2 + p] = skip[i];
  }
}

__global__ void rau_conv1x1_kernel(const float* __restrict__ x,
                                   const float* __restrict__ w,
                                   const float* __restrict__ b,
                                   float* __restrict__ out, int NB, int IC,
                                   int OC, int HW) {
  const long tot = (long)NB * OC * HW;
  for (long i = (long)blockIdx.x * blockDim.x + threadIdx.x; i < tot;
       i += (long)gridDim.x * blockDim.x) {
    long p = i % HW;
    long t = i / HW;
    int oc = (int)(t % OC);
    int nb = (int)(t / OC);
    float acc = b[oc];
    for (int c = 0; c < IC; ++c)
      acc += x[((long)nb * IC + c) * HW + p] * w[oc * IC + c];
    out[i] = acc;
  }
}

// =====================================================================
// Host orchestration
// =====================================================================
extern "C" void kernel_launch(void* const* d_in, const int* in_sizes, int n_in,
                              void* d_out, int out_size, void* d_ws,
                              size_t ws_size, hipStream_t stream) {
  (void)in_sizes; (void)n_in; (void)out_size; (void)ws_size;
  const int NB = 12, REACH = 2, V = NB - 2 * REACH;

  // jax tree-flatten (sorted-key) leaf order
  const float* b1_b  = (const float*)d_in[0];
  const float* b1_bt = (const float*)d_in[1];
  const float* b1_g  = (const float*)d_in[2];
  const float* b1_w  = (const float*)d_in[3];
  const float* d1_c1b = (const float*)d_in[4];
  const float* d1_c1bt= (const float*)d_in[5];
  const float* d1_c1g = (const float*)d_in[6];
  const float* d1_c1w = (const float*)d_in[7];
  const float* d1_upb = (const float*)d_in[8];
  const float* d1_upw = (const float*)d_in[9];
  const float* d1_wk  = (const float*)d_in[10];
  const float* d1_wq  = (const float*)d_in[11];
  const float* d1_wv  = (const float*)d_in[12];
  const float* d2_c1b = (const float*)d_in[13];
  const float* d2_c1bt= (const float*)d_in[14];
  const float* d2_c1g = (const float*)d_in[15];
  const float* d2_c1w = (const float*)d_in[16];
  const float* d2_upb = (const float*)d_in[17];
  const float* d2_upw = (const float*)d_in[18];
  const float* d2_wk  = (const float*)d_in[19];
  const float* d2_wq  = (const float*)d_in[20];
  const float* d2_wv  = (const float*)d_in[21];
  const float* d3_c1b = (const float*)d_in[22];
  const float* d3_c1bt= (const float*)d_in[23];
  const float* d3_c1g = (const float*)d_in[24];
  const float* d3_c1w = (const float*)d_in[25];
  const float* d3_upb = (const float*)d_in[26];
  const float* d3_upw = (const float*)d_in[27];
  const float* d3_wk  = (const float*)d_in[28];
  const float* d3_wq  = (const float*)d_in[29];
  const float* d3_wv  = (const float*)d_in[30];
  const float* e1_b = (const float*)d_in[31];
  const float* e1_bt= (const float*)d_in[32];
  const float* e1_g = (const float*)d_in[33];
  const float* e1_w = (const float*)d_in[34];
  const float* e2_b = (const float*)d_in[35];
  const float* e2_bt= (const float*)d_in[36];
  const float* e2_g = (const float*)d_in[37];
  const float* e2_w = (const float*)d_in[38];
  const float* e3_b = (const float*)d_in[39];
  const float* e3_bt= (const float*)d_in[40];
  const float* e3_g = (const float*)d_in[41];
  const float* e3_w = (const float*)d_in[42];
  const float* out_b = (const float*)d_in[43];
  const float* out_w = (const float*)d_in[44];
  const float* x_in  = (const float*)d_in[45];

  // workspace bump allocator (floats)
  float* ws = (float*)d_ws;
  size_t off = 0;
  auto alloc = [&](size_t n) {
    float* p = ws + off;
    off += (n + 255) & ~(size_t)255;
    return p;
  };
  float* pre  = alloc(7077888);
  float* s1   = alloc(7077888);
  float* p1   = alloc(1769472);
  float* s2   = alloc(3538944);
  float* p2   = alloc(884736);
  float* s3   = alloc(1769472);
  float* p3   = alloc(442368);
  float* b1o  = alloc(884736);
  float* d1o  = alloc(1769472);
  float* d2o  = alloc(3538944);
  float* d3o  = alloc(7077888);
  float* nx   = alloc(3538944);
  float* catb = alloc(14155776);
  float* meanb = alloc(512);
  float* rstdb = alloc(512);
  unsigned short* Qb  = (unsigned short*)alloc(1179648);  // <= 8*2304*128
  unsigned short* Kb  = (unsigned short*)alloc(4718592);  // <= 8*9216*128
  unsigned short* VbT = (unsigned short*)alloc(4718592);

  auto conv_block = [&](const float* in, int IC, int OC, int H, int W,
                        const float* w, const float* b, const float* g,
                        const float* beta, float* outb) {
    int HW = H * W, NP = NB * HW;
    rau_conv3x3_kernel<<<dim3(NP / 64, OC / 32), 256, 0, stream>>>(
        in, w, b, pre, NB, IC, OC, H, W);
    rau_bn_stats_kernel<<<OC, 256, 0, stream>>>(pre, meanb, rstdb, NB, OC, HW);
    long tot = (long)NB * OC * HW;
    rau_bn_relu_kernel<<<(unsigned)((tot + 255) / 256), 256, 0, stream>>>(
        pre, meanb, rstdb, g, beta, outb, OC, HW, tot);
  };

  auto decoder = [&](const float* xdec, const float* skip, int D, int h, int w,
                     int OC, const float* wq, const float* wk, const float* wv,
                     const float* upw, const float* upb, const float* c1w,
                     const float* c1b, const float* c1g, const float* c1bt,
                     float* outb) {
    int L = h * w, M = 4 * L;
    rau_attn_proj_kernel<<<dim3(D / 64, (L + 31) / 32, V), 256, 0, stream>>>(
        xdec, wq, Qb, D, L, L, REACH, 0, 0);
    rau_attn_proj_kernel<<<dim3(D / 64, (M + 31) / 32, V), 256, 0, stream>>>(
        xdec, wk, Kb, D, L, M, REACH, 1, 0);
    rau_attn_proj_kernel<<<dim3(D / 64, (M + 31) / 32, V), 256, 0, stream>>>(
        xdec, wv, VbT, D, L, M, REACH, 1, 1);  // V transposed [v][d][m]
    long xtot = (long)NB * D * L;
    rau_copy_kernel<<<(unsigned)((xtot + 255) / 256), 256, 0, stream>>>(
        xdec, nx, xtot);
    size_t shmem = (size_t)(16 * D + 512 + 48) * 4 + (size_t)(16 * D + 512) * 2;
    rau_flash_attn_kernel<<<dim3(L / 16, V), 128, shmem, stream>>>(
        Qb, Kb, VbT, nx, D, L, M, REACH);
    int NP = NB * L;
    rau_deconv_fuse_kernel<<<dim3(NP / 64, OC / 32, 4), 256, 0, stream>>>(
        nx, upw, upb, skip, catb, NB, D, OC, h, w);
    long stot = (long)NB * OC * 4 * L;
    rau_concat_skip_kernel<<<(unsigned)((stot + 255) / 256), 256, 0, stream>>>(
        skip, catb, NB, OC, (long)4 * L);
    conv_block(catb, 2 * OC, OC, 2 * h, 2 * w, c1w, c1b, c1g, c1bt, outb);
  };

  // encoder
  conv_block(x_in, 3, 64, 96, 96, e1_w, e1_b, e1_g, e1_bt, s1);
  rau_maxpool_kernel<<<(12 * 64 * 48 * 48 + 255) / 256, 256, 0, stream>>>(
      s1, p1, (long)12 * 64, 96, 96);
  conv_block(p1, 64, 128, 48, 48, e2_w, e2_b, e2_g, e2_bt, s2);
  rau_maxpool_kernel<<<(12 * 128 * 24 * 24 + 255) / 256, 256, 0, stream>>>(
      s2, p2, (long)12 * 128, 48, 48);
  conv_block(p2, 128, 256, 24, 24, e3_w, e3_b, e3_g, e3_bt, s3);
  rau_maxpool_kernel<<<(12 * 256 * 12 * 12 + 255) / 256, 256, 0, stream>>>(
      s3, p3, (long)12 * 256, 24, 24);
  conv_block(p3, 256, 512, 12, 12, b1_w, b1_b, b1_g, b1_bt, b1o);

  // decoders
  decoder(b1o, s3, 512, 12, 12, 256, d1_wq, d1_wk, d1_wv, d1_upw, d1_upb,
          d1_c1w, d1_c1b, d1_c1g, d1_c1bt, d1o);
  decoder(d1o, s2, 256, 24, 24, 128, d2_wq, d2_wk, d2_wv, d2_upw, d2_upb,
          d2_c1w, d2_c1b, d2_c1g, d2_c1bt, d2o);
  decoder(d2o, s1, 128, 48, 48, 64, d3_wq, d3_wk, d3_wv, d3_upw, d3_upb,
          d3_c1w, d3_c1b, d3_c1g, d3_c1bt, d3o);

  // head
  long otot = (long)12 * 4 * 96 * 96;
  rau_conv1x1_kernel<<<(unsigned)((otot + 255) / 256), 256, 0, stream>>>(
      d3o, out_w, out_b, (float*)d_out, 12, 64, 4, 96 * 96);
}
